// fusionModule_58239756534203
// MI455X (gfx1250) — matrix operand order repaired and checked by reading, not compiled, not run on hardware
//
#include <hip/hip_runtime.h>
#include <cstdint>

typedef __attribute__((ext_vector_type(16))) _Float16 v16h;
typedef __attribute__((ext_vector_type(8)))  float    v8f;

namespace {
constexpr int Bn = 8, Cn = 256, Hn = 32, Wn = 32, Ln = Hn * Wn, BL = Bn * Ln;
constexpr int HEADS = 4, DH = Cn / HEADS;           // 64
constexpr int DI = 512, DS = 16, DTR = 16, ICn = 64, DCONV = 4;
// tiled gemm geometry
constexpr int BM = 128, BN = 64, BK = 32, LDP = 36;  // LDS pitch (floats), pad 4
}

__device__ __forceinline__ float sigmoidf_(float x) { return 1.f / (1.f + __expf(-x)); }
__device__ __forceinline__ float siluf_(float x)    { return x * sigmoidf_(x); }
__device__ __forceinline__ float softplusf_(float x){ return (x > 20.f) ? x : log1pf(__expf(x)); }

// async copy 16B global -> LDS (per-lane addresses), tracked by ASYNCcnt
__device__ __forceinline__ void async_b128(const float* g, const float* l) {
  unsigned loff = (unsigned)(uintptr_t)l;  // low 32 bits of flat LDS ptr == LDS address
  asm volatile("global_load_async_to_lds_b128 %0, %1, off"
               :: "v"(loff), "v"(g) : "memory");
}
__device__ __forceinline__ void wait_async6() {
  asm volatile("s_wait_asynccnt 0x6" ::: "memory");
}
__device__ __forceinline__ void wait_async0() {
  asm volatile("s_wait_asynccnt 0x0" ::: "memory");
}

__device__ __forceinline__ void cvt4(v16h& a, int base, float4 u) {
  a[base + 0] = (_Float16)u.x; a[base + 1] = (_Float16)u.y;
  a[base + 2] = (_Float16)u.z; a[base + 3] = (_Float16)u.w;
}

// ---------------------------------------------------------------------------
// Tiled GEMM: Y[M,N] = X[M,K] @ W[N,K]^T (+bias). Requires M%128==0, N%64==0,
// K%32==0. 8 waves (4m x 2n), wave tile 32x32 (2x2 WMMA), LDS double-buffered
// via global_load_async_to_lds_b128 (6 async ops / thread / K-step).
// ---------------------------------------------------------------------------
__global__ __launch_bounds__(256)
void k_gemm_tiled(const float* __restrict__ X, int ldx,
                  const float* __restrict__ W, int ldw,
                  const float* __restrict__ bias,
                  float* __restrict__ Y, int ldy,
                  int K, int act) {
  __shared__ __align__(16) float smem[2 * BM * LDP + 2 * BN * LDP];

  const int lane = threadIdx.x, hf = lane >> 4, lq = lane & 15;
  const int wid = threadIdx.y, wm = wid >> 1, wn = wid & 1;
  const int tid = wid * 32 + lane;
  const int m0 = blockIdx.y * BM, n0 = blockIdx.x * BN;
  const int ka = hf * 8, kb = hf * 16;

  auto Asb = [&](int buf) -> float* { return smem + buf * (BM * LDP); };
  auto Bsb = [&](int buf) -> float* { return smem + 2 * BM * LDP + buf * (BN * LDP); };

  auto issue = [&](int buf, int k0) {
    float* As = Asb(buf);
    float* Bs = Bsb(buf);
#pragma unroll
    for (int i = 0; i < 4; ++i) {                 // A: 128 rows x 32 f32
      int ch = tid + i * 256;
      int row = ch >> 3, c16 = ch & 7;
      async_b128(X + (size_t)(m0 + row) * ldx + k0 + c16 * 4,
                 As + row * LDP + c16 * 4);
    }
#pragma unroll
    for (int i = 0; i < 2; ++i) {                 // B: 64 rows x 32 f32
      int ch = tid + i * 256;
      int row = ch >> 3, c16 = ch & 7;
      async_b128(W + (size_t)(n0 + row) * ldw + k0 + c16 * 4,
                 Bs + row * LDP + c16 * 4);
    }
  };

  v8f acc[2][2] = {};
  const int nk = K / BK;
  issue(0, 0);
  for (int kk = 0; kk < nk; ++kk) {
    const int cur = kk & 1;
    if (kk + 1 < nk) { issue(cur ^ 1, (kk + 1) * BK); wait_async6(); }
    else             { wait_async0(); }
    __syncthreads();
    const float* As = Asb(cur);
    const float* Bs = Bsb(cur);

    // Phase 1: batch all LDS loads into independent temporaries (one clause).
    float4 ta[2][4], tb[2][4];
#pragma unroll
    for (int mi = 0; mi < 2; ++mi) {
      const float* ar = As + (wm * 32 + mi * 16 + lq) * LDP;
      ta[mi][0] = *(const float4*)(ar + ka);
      ta[mi][1] = *(const float4*)(ar + ka + 4);
      ta[mi][2] = *(const float4*)(ar + 16 + ka);
      ta[mi][3] = *(const float4*)(ar + 16 + ka + 4);
    }
#pragma unroll
    for (int ni = 0; ni < 2; ++ni) {
      const float* br = Bs + (wn * 32 + ni * 16 + lq) * LDP + ka;
      tb[ni][0] = *(const float4*)(br);
      tb[ni][1] = *(const float4*)(br + 4);
      tb[ni][2] = *(const float4*)(br + 16);
      tb[ni][3] = *(const float4*)(br + 20);
    }

    // Phase 2: convert to f16 fragments.
    v16h a[2], b[2];
#pragma unroll
    for (int mi = 0; mi < 2; ++mi)
#pragma unroll
      for (int q = 0; q < 4; ++q) cvt4(a[mi], q * 4, ta[mi][q]);
#pragma unroll
    for (int ni = 0; ni < 2; ++ni)
#pragma unroll
      for (int q = 0; q < 4; ++q) cvt4(b[ni], q * 4, tb[ni][q]);

#pragma unroll
    for (int mi = 0; mi < 2; ++mi)
#pragma unroll
      for (int ni = 0; ni < 2; ++ni)
        acc[mi][ni] = __builtin_amdgcn_wmma_f32_16x16x32_f16(
            false, a[mi], false, b[ni], (short)0, acc[mi][ni], false, false);
    __syncthreads();
  }

#pragma unroll
  for (int ni = 0; ni < 2; ++ni) {
    const int col = n0 + wn * 32 + ni * 16 + lq;
    const float bv = bias ? bias[col] : 0.f;
#pragma unroll
    for (int mi = 0; mi < 2; ++mi) {
      float* yr = Y + (size_t)(m0 + wm * 32 + mi * 16 + hf * 8) * ldy + col;
#pragma unroll
      for (int r = 0; r < 8; ++r) {
        float v = acc[mi][ni][r] + bv;
        if (act == 1) v = softplusf_(v);
        yr[(size_t)r * ldy] = v;
      }
    }
  }
}

// ---------------------------------------------------------------------------
// Direct WMMA GEMM for odd shapes (N%64!=0 or K%32!=0). One wave per 16x16.
// ---------------------------------------------------------------------------
template <bool TAIL>
__global__ __launch_bounds__(128)
void k_gemm_xwT(const float* __restrict__ X, int ldx,
                const float* __restrict__ W, int ldw,
                const float* __restrict__ bias,
                float* __restrict__ Y, int ldy,
                int K, int act) {
  const int lane = threadIdx.x;
  const int hf = lane >> 4;
  const int lq = lane & 15;
  const int m0 = (blockIdx.y * blockDim.y + threadIdx.y) * 16;
  const int n0 = blockIdx.x * 16;
  const float* xr = X + (size_t)(m0 + lq) * ldx;
  const float* wr = W + (size_t)(n0 + lq) * ldw;
  const int ka = hf * 8, kb = hf * 16;
  v8f acc = {};
  for (int k0 = 0; k0 < K; k0 += 32) {
    v16h a, b;
#pragma unroll
    for (int j = 0; j < 8; ++j) {
      int k1 = k0 + ka + j, k2 = k0 + 16 + ka + j;
      if (TAIL) {
        a[j]     = (_Float16)((k1 < K) ? xr[k1] : 0.f);
        a[8 + j] = (_Float16)((k2 < K) ? xr[k2] : 0.f);
      } else {
        a[j]     = (_Float16)xr[k1];
        a[8 + j] = (_Float16)xr[k2];
      }
    }
#pragma unroll
    for (int j = 0; j < 16; ++j) {
      int kk = k0 + ka + j + (j & 8);
      if (TAIL) b[j] = (_Float16)((kk < K) ? wr[kk] : 0.f);
      else      b[j] = (_Float16)wr[kk];
    }
    acc = __builtin_amdgcn_wmma_f32_16x16x32_f16(false, a, false, b, (short)0, acc, false, false);
  }
  const int col = n0 + lq;
  const float bv = bias ? bias[col] : 0.f;
  float* yr = Y + (size_t)(m0 + hf * 8) * ldy + col;
#pragma unroll
  for (int r = 0; r < 8; ++r) {
    float v = acc[r] + bv;
    if (act == 1) v = softplusf_(v);
    yr[(size_t)r * ldy] = v;
  }
}

// ---------------------------------------------------------------------------
// Flash attention, one wave per (b, head, 16-row q tile). dh=64, scale=1/8.
// ---------------------------------------------------------------------------
__global__ __launch_bounds__(32)
void k_flash(const float* __restrict__ Q, const float* __restrict__ Km,
             const float* __restrict__ V, float* __restrict__ O) {
  __shared__ float sS[16 * 32];
  __shared__ float sM[16], sL[16], sA[16];
  const int lane = threadIdx.x, hf = lane >> 4, lq = lane & 15;
  const int q0 = blockIdx.x * 16, hd = blockIdx.y, b = blockIdx.z;
  const size_t base = (size_t)b * Ln * Cn + (size_t)hd * DH;
  const float* Qb = Q + base;
  const float* Kb = Km + base;
  const float* Vb = V + base;

  v16h aq[2];
  {
    const float* qr = Qb + (size_t)(q0 + lq) * Cn;
#pragma unroll
    for (int j = 0; j < 8; ++j) {
      aq[0][j]     = (_Float16)qr[hf * 8 + j];
      aq[0][8 + j] = (_Float16)qr[16 + hf * 8 + j];
      aq[1][j]     = (_Float16)qr[32 + hf * 8 + j];
      aq[1][8 + j] = (_Float16)qr[48 + hf * 8 + j];
    }
  }
  v8f o[4] = {};
  if (lane < 16) { sM[lane] = -3.0e38f; sL[lane] = 0.f; }
  __syncthreads();

  for (int kt = 0; kt < Ln; kt += 32) {
    v8f s0 = {}, s1 = {};
#pragma unroll
    for (int kc = 0; kc < 2; ++kc) {
      v16h bk;
      const float* kr = Kb + (size_t)(kt + lq) * Cn + kc * 32 + hf * 8;
#pragma unroll
      for (int j = 0; j < 16; ++j) bk[j] = (_Float16)kr[j + (j & 8)];
      s0 = __builtin_amdgcn_wmma_f32_16x16x32_f16(false, aq[kc], false, bk, (short)0, s0, false, false);
    }
#pragma unroll
    for (int kc = 0; kc < 2; ++kc) {
      v16h bk;
      const float* kr = Kb + (size_t)(kt + 16 + lq) * Cn + kc * 32 + hf * 8;
#pragma unroll
      for (int j = 0; j < 16; ++j) bk[j] = (_Float16)kr[j + (j & 8)];
      s1 = __builtin_amdgcn_wmma_f32_16x16x32_f16(false, aq[kc], false, bk, (short)0, s1, false, false);
    }
#pragma unroll
    for (int r = 0; r < 8; ++r) {
      sS[(hf * 8 + r) * 32 + lq]      = s0[r] * 0.125f;
      sS[(hf * 8 + r) * 32 + 16 + lq] = s1[r] * 0.125f;
    }
    __syncthreads();
    if (lane < 16) {
      float mo = sM[lane], mn = mo;
      for (int j = 0; j < 32; ++j) mn = fmaxf(mn, sS[lane * 32 + j]);
      float al = __expf(mo - mn), sum = 0.f;
      for (int j = 0; j < 32; ++j) {
        float p = __expf(sS[lane * 32 + j] - mn);
        sS[lane * 32 + j] = p;
        sum += p;
      }
      sM[lane] = mn;
      sL[lane] = sL[lane] * al + sum;
      sA[lane] = al;
    }
    __syncthreads();
#pragma unroll
    for (int r = 0; r < 8; ++r) {
      float al = sA[hf * 8 + r];
#pragma unroll
      for (int c = 0; c < 4; ++c) o[c][r] *= al;
    }
    v16h ap;
#pragma unroll
    for (int j = 0; j < 8; ++j) {
      ap[j]     = (_Float16)sS[lq * 32 + hf * 8 + j];
      ap[8 + j] = (_Float16)sS[lq * 32 + 16 + hf * 8 + j];
    }
    const float* vb0 = Vb + (size_t)(kt + hf * 8) * Cn + lq;
#pragma unroll
    for (int c = 0; c < 4; ++c) {
      v16h bv;
#pragma unroll
      for (int j = 0; j < 16; ++j) bv[j] = (_Float16)vb0[(size_t)(j + (j & 8)) * Cn + c * 16];
      o[c] = __builtin_amdgcn_wmma_f32_16x16x32_f16(false, ap, false, bv, (short)0, o[c], false, false);
    }
    __syncthreads();
  }

  float* Ob = O + base;
#pragma unroll
  for (int r = 0; r < 8; ++r) {
    float inv = 1.f / sL[hf * 8 + r];
    float* orow = Ob + (size_t)(q0 + hf * 8 + r) * Cn + lq;
#pragma unroll
    for (int c = 0; c < 4; ++c) orow[c * 16] = o[c][r] * inv;
  }
}

// ---------------------------------------------------------------------------
// Criss-cross attention branch: dir=0 horizontal, dir=1 vertical.
// ---------------------------------------------------------------------------
__global__ __launch_bounds__(32)
void k_cca(const float* __restrict__ cq, const float* __restrict__ ck,
           const float* __restrict__ cv, float* __restrict__ out, int dir) {
  __shared__ float sk[32 * ICn];
  __shared__ float sv[32 * Cn];
  const int t = threadIdx.x;
  const int fx = blockIdx.x, b = blockIdx.y;
  auto lidx = [&](int i) { return dir ? (i * Wn + fx) : (fx * Wn + i); };
  for (int idx = t; idx < 32 * ICn; idx += 32) {
    int i = idx / ICn, c = idx % ICn;
    sk[idx] = ck[((size_t)b * Ln + lidx(i)) * ICn + c];
  }
  for (int idx = t; idx < 32 * Cn; idx += 32) {
    int i = idx / Cn, c = idx % Cn;
    sv[idx] = cv[((size_t)b * Ln + lidx(i)) * Cn + c];
  }
  __syncthreads();
  float qreg[ICn];
  const float* qr = cq + ((size_t)b * Ln + lidx(t)) * ICn;
#pragma unroll
  for (int c = 0; c < ICn; ++c) qreg[c] = qr[c];
  float sc[32];
  float mx = -3e38f;
  for (int k = 0; k < 32; ++k) {
    float s = 0.f;
#pragma unroll
    for (int c = 0; c < ICn; ++c) s += qreg[c] * sk[k * ICn + c];
    sc[k] = s;
    mx = fmaxf(mx, s);
  }
  float sum = 0.f;
  for (int k = 0; k < 32; ++k) { sc[k] = __expf(sc[k] - mx); sum += sc[k]; }
  float inv = 1.f / sum;
  float* orow = out + ((size_t)b * Ln + lidx(t)) * Cn;
  for (int c = 0; c < Cn; ++c) {
    float acc = 0.f;
    for (int k = 0; k < 32; ++k) acc += sc[k] * sv[k * Cn + c];
    orow[c] = acc * inv;
  }
}

__global__ void k_combine(const float* __restrict__ x, const float* __restrict__ oh,
                          const float* __restrict__ ov, const float* __restrict__ gamma,
                          float* __restrict__ out, int n) {
  int i = blockIdx.x * blockDim.x + threadIdx.x;
  if (i < n) out[i] = gamma[0] * (oh[i] + ov[i]) + x[i];
}

__global__ void k_cl2lc(const float* __restrict__ in, float* __restrict__ out) {
  int i = blockIdx.x * blockDim.x + threadIdx.x;
  if (i >= Bn * Ln * Cn) return;
  int c = i % Cn, l = (i / Cn) % Ln, b = i / (Cn * Ln);
  out[i] = in[((size_t)b * Cn + c) * Ln + l];
}

__global__ void k_lc2cl(const float* __restrict__ in, float* __restrict__ out) {
  int i = blockIdx.x * blockDim.x + threadIdx.x;
  if (i >= Bn * Cn * Ln) return;
  int l = i % Ln, c = (i / Ln) % Cn, b = i / (Ln * Cn);
  out[i] = in[((size_t)b * Ln + l) * Cn + c];
}

// causal depthwise conv over first DI channels of xz (B,L,2*DI), +bias, SiLU.
__global__ void k_conv_silu(const float* __restrict__ xz, const float* __restrict__ cw,
                            const float* __restrict__ cb, float* __restrict__ u) {
  int i = blockIdx.x * blockDim.x + threadIdx.x;
  if (i >= Bn * Ln * DI) return;
  int d = i % DI, l = (i / DI) % Ln, b = i / (DI * Ln);
  float acc = cb[d];
#pragma unroll
  for (int j = 0; j < DCONV; ++j) {
    int lt = l - (DCONV - 1) + j;
    if (lt >= 0) acc += cw[d * DCONV + j] * xz[((size_t)b * Ln + lt) * (2 * DI) + d];
  }
  u[i] = siluf_(acc);
}

// selective scan: one thread per (b,d); 16 states in registers, loop over L.
__global__ __launch_bounds__(256)
void k_scan(const float* __restrict__ delta, const float* __restrict__ u,
            const float* __restrict__ xdbl, const float* __restrict__ xz,
            const float* __restrict__ A_log, const float* __restrict__ Dp,
            float* __restrict__ y) {
  int i = blockIdx.x * blockDim.x + threadIdx.x;
  if (i >= Bn * DI) return;
  int d = i % DI, b = i / DI;
  float Ar[DS], hs[DS];
#pragma unroll
  for (int n = 0; n < DS; ++n) { Ar[n] = -__expf(A_log[d * DS + n]); hs[n] = 0.f; }
  const float dpv = Dp[d];
  for (int t = 0; t < Ln; ++t) {
    size_t rb = (size_t)b * Ln + t;
    float dl = delta[rb * DI + d];
    float uu = u[rb * DI + d];
    const float* xd = xdbl + rb * 48;
    float yv = 0.f;
#pragma unroll
    for (int n = 0; n < DS; ++n) {
      float dA = __expf(dl * Ar[n]);
      hs[n] = dA * hs[n] + dl * xd[16 + n] * uu;
      yv += hs[n] * xd[32 + n];
    }
    yv += uu * dpv;
    float zz = xz[rb * (2 * DI) + DI + d];
    y[rb * DI + d] = yv * siluf_(zz);
  }
}

// ---------------------------------------------------------------------------
extern "C" void kernel_launch(void* const* d_in, const int* in_sizes, int n_in,
                              void* d_out, int out_size, void* d_ws, size_t ws_size,
                              hipStream_t stream) {
  (void)in_sizes; (void)n_in; (void)out_size; (void)ws_size;
  const float* sam = (const float*)d_in[0];
  const float* myf = (const float*)d_in[1];
  const float* wq  = (const float*)d_in[2];  const float* bq  = (const float*)d_in[3];
  const float* wk  = (const float*)d_in[4];  const float* bk  = (const float*)d_in[5];
  const float* wv  = (const float*)d_in[6];  const float* bv  = (const float*)d_in[7];
  const float* wo  = (const float*)d_in[8];  const float* bo  = (const float*)d_in[9];
  const float* cqw = (const float*)d_in[10]; const float* cqb = (const float*)d_in[11];
  const float* ckw = (const float*)d_in[12]; const float* ckb = (const float*)d_in[13];
  const float* cvw = (const float*)d_in[14]; const float* cvb = (const float*)d_in[15];
  const float* gamma = (const float*)d_in[16];
  const float* tow = (const float*)d_in[17]; const float* tob = (const float*)d_in[18];
  const float* ipw = (const float*)d_in[19];
  const float* cw  = (const float*)d_in[20]; const float* cb  = (const float*)d_in[21];
  const float* xpw = (const float*)d_in[22];
  const float* dtw = (const float*)d_in[23]; const float* dtb = (const float*)d_in[24];
  const float* Alog = (const float*)d_in[25]; const float* Dp = (const float*)d_in[26];
  const float* opw = (const float*)d_in[27];
  const float* fsw = (const float*)d_in[28]; const float* fsb = (const float*)d_in[29];
  float* out = (float*)d_out;

  float* f = (float*)d_ws;
  const size_t S2 = (size_t)BL * Cn;            // 2M floats
  float* w0 = f; f += S2;
  float* w1 = f; f += S2;
  float* w2 = f; f += S2;
  float* w3 = f; f += S2;
  float* w4 = f; f += S2;
  float* w5 = f; f += S2;
  float* wbig = f; f += (size_t)BL * 2 * DI;    // 8M floats (xz)
  float* wd0 = f; f += (size_t)BL * DI;         // u
  float* wd1 = f; f += (size_t)BL * DI;         // delta
  float* wd2 = f; f += (size_t)BL * DI;         // y

  auto G = [&](const float* X, int ldx, const float* W_, int ldw, const float* bias,
               float* Y, int ldy, int M, int N, int K, int act) {
    if ((N % BN == 0) && (K % BK == 0) && (M % BM == 0)) {
      dim3 blk(32, 8, 1), grd(N / BN, M / BM, 1);
      k_gemm_tiled<<<grd, blk, 0, stream>>>(X, ldx, W_, ldw, bias, Y, ldy, K, act);
    } else {
      dim3 blk(32, 4, 1), grd(N / 16, M / 64, 1);
      if (K % 32 == 0)
        k_gemm_xwT<false><<<grd, blk, 0, stream>>>(X, ldx, W_, ldw, bias, Y, ldy, K, act);
      else
        k_gemm_xwT<true><<<grd, blk, 0, stream>>>(X, ldx, W_, ldw, bias, Y, ldy, K, act);
    }
  };

  const int EL2 = BL * Cn;
  k_cl2lc<<<(EL2 + 255) / 256, 256, 0, stream>>>(sam, w0);
  k_cl2lc<<<(EL2 + 255) / 256, 256, 0, stream>>>(myf, w1);

  // cross attention
  G(w0, Cn, wq, Cn, bq, w2, Cn, BL, Cn, Cn, 0);               // Q
  G(w1, Cn, wk, Cn, bk, w3, Cn, BL, Cn, Cn, 0);               // K
  G(w1, Cn, wv, Cn, bv, w4, Cn, BL, Cn, Cn, 0);               // V
  k_flash<<<dim3(Ln / 16, HEADS, Bn), 32, 0, stream>>>(w2, w3, w4, w5);
  G(w5, Cn, wo, Cn, bo, w0, Cn, BL, Cn, Cn, 0);               // fusion

  // criss-cross attention
  G(w0, Cn, cqw, Cn, cqb, w2, ICn, BL, ICn, Cn, 0);           // cq
  G(w0, Cn, ckw, Cn, ckb, w3, ICn, BL, ICn, Cn, 0);           // ck
  G(w0, Cn, cvw, Cn, cvb, w4, Cn, BL, Cn, Cn, 0);             // cv
  k_cca<<<dim3(Hn, Bn), 32, 0, stream>>>(w2, w3, w4, w5, 0);  // out_h
  k_cca<<<dim3(Wn, Bn), 32, 0, stream>>>(w2, w3, w4, w1, 1);  // out_v
  k_combine<<<(EL2 + 255) / 256, 256, 0, stream>>>(w0, w5, w1, gamma, w2, EL2);

  // mamba blocks
  for (int i = 0; i < 2; ++i) {
    G(w2, Cn, tow + (size_t)i * Cn * Cn, Cn, tob + (size_t)i * Cn, w3, Cn, BL, Cn, Cn, 0);
    G(w3, Cn, ipw + (size_t)i * 2 * DI * Cn, Cn, nullptr, wbig, 2 * DI, BL, 2 * DI, Cn, 0);
    int nconv = BL * DI;
    k_conv_silu<<<(nconv + 255) / 256, 256, 0, stream>>>(
        wbig, cw + (size_t)i * DI * DCONV, cb + (size_t)i * DI, wd0);
    G(wd0, DI, xpw + (size_t)i * 48 * DI, DI, nullptr, w4, 48, BL, 48, DI, 0);
    G(w4, 48, dtw + (size_t)i * DI * DTR, DTR, dtb + (size_t)i * DI, wd1, DI, BL, DI, DTR, 1);
    k_scan<<<(Bn * DI + 255) / 256, 256, 0, stream>>>(
        wd1, wd0, w4, wbig, Alog + (size_t)i * DI * DS, Dp + (size_t)i * DI, wd2);
    G(wd2, DI, opw + (size_t)i * Cn * DI, DI, nullptr, w5, Cn, BL, Cn, DI, 0);
    G(w5, Cn, fsw + (size_t)i * Cn * Cn, Cn, fsb + (size_t)i * Cn, w2, Cn, BL, Cn, Cn, 0);
  }

  k_lc2cl<<<(EL2 + 255) / 256, 256, 0, stream>>>(w2, out);
}
